// GCN_Layer_16509854285891
// MI455X (gfx1250) — matrix operand order repaired and checked by reading, not compiled, not run on hardware
//
#include <hip/hip_runtime.h>
#include <hip/hip_bf16.h>

#define D_FEAT 256

typedef _Float16 v16h __attribute__((ext_vector_type(16)));
typedef float    v8f  __attribute__((ext_vector_type(8)));

// ---------------------------------------------------------------------------
// Degree counting / normalization
// ---------------------------------------------------------------------------
__global__ void k_zero2(unsigned* __restrict__ a, unsigned* __restrict__ b, int n) {
  int i = blockIdx.x * blockDim.x + threadIdx.x;
  if (i < n) { a[i] = 0u; b[i] = 0u; }
}

__global__ void k_count_deg(const long long* __restrict__ edst,
                            unsigned* __restrict__ cnt, int nedges) {
  int e = blockIdx.x * blockDim.x + threadIdx.x;
  if (e < nedges) atomicAdd(&cnt[(int)edst[e]], 1u);
}

__global__ void k_dinv(const unsigned* __restrict__ cnt,
                       float* __restrict__ dinv, int n) {
  int i = blockIdx.x * blockDim.x + threadIdx.x;
  if (i < n) dinv[i] = rsqrtf((float)cnt[i] + 1.0f);
}

// ---------------------------------------------------------------------------
// Single-block exclusive scan of cnt[0..n) -> offsets[0..n).
// 1024 threads; each owns a contiguous chunk; Hillis-Steele over partials.
// ---------------------------------------------------------------------------
__global__ void k_scan(const unsigned* __restrict__ cnt,
                       unsigned* __restrict__ offsets, int n) {
  __shared__ unsigned buf[2][1024];
  const int t = threadIdx.x;
  const int chunk = (n + 1023) / 1024;
  const int base = t * chunk;

  unsigned partial = 0u;
  for (int i = 0; i < chunk; ++i) {
    int idx = base + i;
    if (idx < n) partial += cnt[idx];
  }
  buf[0][t] = partial;
  __syncthreads();

  int cur = 0;
  for (int d = 1; d < 1024; d <<= 1) {
    unsigned v = buf[cur][t];
    if (t >= d) v += buf[cur][t - d];
    buf[1 - cur][t] = v;
    cur = 1 - cur;
    __syncthreads();
  }
  unsigned running = buf[cur][t] - partial;   // exclusive prefix of this chunk
  for (int i = 0; i < chunk; ++i) {
    int idx = base + i;
    if (idx < n) { offsets[idx] = running; running += cnt[idx]; }
  }
}

// ---------------------------------------------------------------------------
// Bucket edges into CSR order (u32 cursor atomics only).
// ---------------------------------------------------------------------------
__global__ void k_bucket(const long long* __restrict__ esrc,
                         const long long* __restrict__ edst,
                         const unsigned* __restrict__ offsets,
                         unsigned* __restrict__ fill,
                         const float* __restrict__ dinv,
                         int* __restrict__ ssrc, float* __restrict__ scoef,
                         int nedges) {
  int e = blockIdx.x * blockDim.x + threadIdx.x;
  if (e >= nedges) return;
  int s = (int)esrc[e];
  int d = (int)edst[e];
  unsigned p = offsets[d] + atomicAdd(&fill[d], 1u);
  ssrc[p]  = s;
  scoef[p] = dinv[s] * dinv[d];
}

// ---------------------------------------------------------------------------
// Transpose + f16-convert weights: WT[col][k] = (f16) W[k][col]
// ---------------------------------------------------------------------------
__global__ void k_wt(const float* __restrict__ W, _Float16* __restrict__ WT) {
  int i = blockIdx.x * blockDim.x + threadIdx.x;   // over 256*256
  int c = i >> 8;
  int k = i & 255;
  WT[i] = (_Float16)W[(size_t)k * D_FEAT + c];
}

// ---------------------------------------------------------------------------
// WMMA GEMM: XW[n,256] = act(A[n,256]) @ W[256,256], W given as f16 WT[col][k].
// Block = 16 rows x 256 cols, 8 waves, each wave owns two 16-col tiles.
// A frag (f16 16x32): a[i]   = A[row][kk + hh*16 + i]       (i<8)
//                     a[8+i] = A[row][kk + hh*16 + 8 + i]
// B frag            : b[i]   = WT[col][kk + hh*16 + i]       (contiguous 32B)
// C/D: VGPR j -> row M = j + 8*hh, col N = lane&15.
// ---------------------------------------------------------------------------
__global__ void k_gemm_wmma(const float* __restrict__ A,
                            const _Float16* __restrict__ WT,
                            float* __restrict__ XW,
                            int nrows, int relu_in) {
  const int r0   = blockIdx.x * 16;
  const int wave = threadIdx.x >> 5;
  const int lane = threadIdx.x & 31;
  const int hh   = lane >> 4;
  const int mrow = lane & 15;
  const int ncol = lane & 15;

  const int c0 = (wave * 2) * 16;
  const int c1 = c0 + 16;

  int arow = r0 + mrow;
  if (arow >= nrows) arow = nrows - 1;
  const float*    __restrict__ Arow = A  + (size_t)arow * D_FEAT;
  const _Float16* __restrict__ wt0  = WT + (size_t)(c0 + ncol) * D_FEAT;
  const _Float16* __restrict__ wt1  = WT + (size_t)(c1 + ncol) * D_FEAT;

  v8f acc0 = {};
  v8f acc1 = {};

  for (int kk = 0; kk < D_FEAT; kk += 32) {
    const float4* pa = (const float4*)(Arow + kk + hh * 16);
    float4 f0 = pa[0];
    float4 f1 = pa[1];
    const float4* pb = (const float4*)(Arow + kk + hh * 16 + 8);
    float4 f2 = pb[0];
    float4 f3 = pb[1];
    if (relu_in) {
      f0.x = fmaxf(f0.x, 0.f); f0.y = fmaxf(f0.y, 0.f); f0.z = fmaxf(f0.z, 0.f); f0.w = fmaxf(f0.w, 0.f);
      f1.x = fmaxf(f1.x, 0.f); f1.y = fmaxf(f1.y, 0.f); f1.z = fmaxf(f1.z, 0.f); f1.w = fmaxf(f1.w, 0.f);
      f2.x = fmaxf(f2.x, 0.f); f2.y = fmaxf(f2.y, 0.f); f2.z = fmaxf(f2.z, 0.f); f2.w = fmaxf(f2.w, 0.f);
      f3.x = fmaxf(f3.x, 0.f); f3.y = fmaxf(f3.y, 0.f); f3.z = fmaxf(f3.z, 0.f); f3.w = fmaxf(f3.w, 0.f);
    }
    v16h a;
    a[0]  = (_Float16)f0.x; a[1]  = (_Float16)f0.y; a[2]  = (_Float16)f0.z; a[3]  = (_Float16)f0.w;
    a[4]  = (_Float16)f1.x; a[5]  = (_Float16)f1.y; a[6]  = (_Float16)f1.z; a[7]  = (_Float16)f1.w;
    a[8]  = (_Float16)f2.x; a[9]  = (_Float16)f2.y; a[10] = (_Float16)f2.z; a[11] = (_Float16)f2.w;
    a[12] = (_Float16)f3.x; a[13] = (_Float16)f3.y; a[14] = (_Float16)f3.z; a[15] = (_Float16)f3.w;

    v16h b0 = *(const v16h*)(wt0 + kk + hh * 16);
    v16h b1 = *(const v16h*)(wt1 + kk + hh * 16);

    acc0 = __builtin_amdgcn_wmma_f32_16x16x32_f16(false, a, false, b0,
                                                  (short)0, acc0, false, false);
    acc1 = __builtin_amdgcn_wmma_f32_16x16x32_f16(false, a, false, b1,
                                                  (short)0, acc1, false, false);
  }

  if (r0 + 16 <= nrows) {
#pragma unroll
    for (int j = 0; j < 8; ++j) {
      int row = r0 + j + hh * 8;
      XW[(size_t)row * D_FEAT + c0 + ncol] = acc0[j];
      XW[(size_t)row * D_FEAT + c1 + ncol] = acc1[j];
    }
  } else {
#pragma unroll
    for (int j = 0; j < 8; ++j) {
      int row = r0 + j + hh * 8;
      if (row < nrows) {
        XW[(size_t)row * D_FEAT + c0 + ncol] = acc0[j];
        XW[(size_t)row * D_FEAT + c1 + ncol] = acc1[j];
      }
    }
  }
}

// ---------------------------------------------------------------------------
// CSR aggregation, one wave per node (no fp atomics):
// out[n] = sum_e coef[e]*xw[src(e)] + dinv[n]^2*xw[n] + bias
// ---------------------------------------------------------------------------
__global__ void k_aggregate(const float* __restrict__ XW,
                            const unsigned* __restrict__ offsets,
                            const unsigned* __restrict__ cnt,
                            const int* __restrict__ ssrc,
                            const float* __restrict__ scoef,
                            const float* __restrict__ dinv,
                            const float* __restrict__ bias,
                            float* __restrict__ out, int n) {
  int node = (int)((blockIdx.x * blockDim.x + threadIdx.x) >> 5);
  if (node >= n) return;
  int lane = threadIdx.x & 31;

  unsigned beg = offsets[node];
  unsigned num = cnt[node];

  float4 a0 = {0.f, 0.f, 0.f, 0.f};
  float4 a1 = {0.f, 0.f, 0.f, 0.f};
  for (unsigned i = 0; i < num; ++i) {
    int s   = ssrc[beg + i];
    float c = scoef[beg + i];
    const float4* p = (const float4*)(XW + (size_t)s * D_FEAT + lane * 8);
    float4 v0 = p[0];
    float4 v1 = p[1];
    a0.x += c * v0.x; a0.y += c * v0.y; a0.z += c * v0.z; a0.w += c * v0.w;
    a1.x += c * v1.x; a1.y += c * v1.y; a1.z += c * v1.z; a1.w += c * v1.w;
  }

  float di = dinv[node];
  float d2 = di * di;
  const float4* ps = (const float4*)(XW + (size_t)node * D_FEAT + lane * 8);
  float4 s0 = ps[0];
  float4 s1 = ps[1];
  const float4* pb = (const float4*)(bias + lane * 8);
  float4 b0 = pb[0];
  float4 b1 = pb[1];

  a0.x += d2 * s0.x + b0.x; a0.y += d2 * s0.y + b0.y;
  a0.z += d2 * s0.z + b0.z; a0.w += d2 * s0.w + b0.w;
  a1.x += d2 * s1.x + b1.x; a1.y += d2 * s1.y + b1.y;
  a1.z += d2 * s1.z + b1.z; a1.w += d2 * s1.w + b1.w;

  float4* po = (float4*)(out + (size_t)node * D_FEAT + lane * 8);
  po[0] = a0;
  po[1] = a1;
}

// ---------------------------------------------------------------------------
static inline size_t align_up(size_t v, size_t a) { return (v + a - 1) & ~(a - 1); }

extern "C" void kernel_launch(void* const* d_in, const int* in_sizes, int n_in,
                              void* d_out, int out_size, void* d_ws, size_t ws_size,
                              hipStream_t stream) {
  const float*     x  = (const float*)d_in[0];
  const long long* ei = (const long long*)d_in[1];
  const float* Wm[3] = { (const float*)d_in[2], (const float*)d_in[4], (const float*)d_in[6] };
  const float* bv[3] = { (const float*)d_in[3], (const float*)d_in[5], (const float*)d_in[7] };
  float* out = (float*)d_out;

  const int n = in_sizes[0] / D_FEAT;       // 100000
  const int E = in_sizes[1] / 2;            // 1600000
  const long long* esrc = ei;
  const long long* edst = ei + E;

  // workspace carve-out
  char* ws = (char*)d_ws;
  size_t off = 0;
  unsigned* cnt     = (unsigned*)(ws + off); off = align_up(off + (size_t)n * 4, 256);
  unsigned* offsets = (unsigned*)(ws + off); off = align_up(off + (size_t)n * 4, 256);
  unsigned* fill    = (unsigned*)(ws + off); off = align_up(off + (size_t)n * 4, 256);
  float*    dinv    = (float*)(ws + off);    off = align_up(off + (size_t)n * 4, 256);
  int*      ssrc    = (int*)(ws + off);      off = align_up(off + (size_t)E * 4, 256);
  float*    scoef   = (float*)(ws + off);    off = align_up(off + (size_t)E * 4, 256);
  _Float16* WT[3];
  for (int l = 0; l < 3; ++l) {
    WT[l] = (_Float16*)(ws + off);           off = align_up(off + (size_t)D_FEAT * D_FEAT * 2, 256);
  }
  float* XW = (float*)(ws + off);            off = align_up(off + (size_t)n * D_FEAT * 4, 256);
  (void)ws_size;

  const int T = 256;

  // ---- preprocessing (shared by all 3 layers) ----
  k_zero2    <<<(n + T - 1) / T, T, 0, stream>>>(cnt, fill, n);
  k_count_deg<<<(E + T - 1) / T, T, 0, stream>>>(edst, cnt, E);
  k_dinv     <<<(n + T - 1) / T, T, 0, stream>>>(cnt, dinv, n);
  k_scan     <<<1, 1024, 0, stream>>>(cnt, offsets, n);
  k_bucket   <<<(E + T - 1) / T, T, 0, stream>>>(esrc, edst, offsets, fill, dinv,
                                                 ssrc, scoef, E);
  for (int l = 0; l < 3; ++l)
    k_wt<<<(D_FEAT * D_FEAT + T - 1) / T, T, 0, stream>>>(Wm[l], WT[l]);

  const int gemm_blocks = (n + 15) / 16;
  const int agg_blocks  = (n * 32 + T - 1) / T;   // one wave per node

  for (int layer = 0; layer < 3; ++layer) {
    const float* h = (layer == 0) ? x : out;   // ReLU fused into GEMM load
    k_gemm_wmma<<<gemm_blocks, T, 0, stream>>>(h, WT[layer], XW, n, layer > 0 ? 1 : 0);
    k_aggregate<<<agg_blocks, T, 0, stream>>>(XW, offsets, cnt, ssrc, scoef,
                                              dinv, bv[layer], out, n);
  }
}
